// Block_26164940767966
// MI455X (gfx1250) — matrix operand order repaired and checked by reading, not compiled, not run on hardware
//
#include <hip/hip_runtime.h>
#include <math.h>

// ---------------------------------------------------------------------------
// CDNA5 (gfx1250) transformer block forward.
// GEMMs/attention use V_WMMA_F32_16X16X32_BF16 (fp32 accumulate).
// Tile staging uses GLOBAL_LOAD_ASYNC_TO_LDS_B128 (ASYNCcnt), double-buffered
// in the GEMM so global latency overlaps the WMMA stream within each wave.
// ---------------------------------------------------------------------------

typedef unsigned short u16;
typedef __attribute__((ext_vector_type(16))) __bf16        v16bf;
typedef __attribute__((ext_vector_type(8)))  float         v8f;
typedef __attribute__((ext_vector_type(4)))  unsigned int  v4u;

union Frag16 {            // 16 bf16 values = 8 VGPRs (one WMMA A or B fragment)
    v16bf bf;
    u16   u[16];
    v4u   q[2];
};

__device__ __forceinline__ u16 f2bf(float f) {
    union { float f; unsigned u; } v; v.f = f;
    unsigned r = v.u + 0x7FFFu + ((v.u >> 16) & 1u);   // round-to-nearest-even
    return (u16)(r >> 16);
}

__device__ __forceinline__ v8f wmma_bf16(const Frag16& a, const Frag16& b, v8f c) {
    return __builtin_amdgcn_wmma_f32_16x16x32_bf16(false, a.bf, false, b.bf,
                                                   (short)0, c, false, false);
}

// Async global->LDS 16B copy (per-lane). LDS dest offset = low 32 bits of
// generic pointer (flat aperture keeps LDS byte offset in addr[31:0]).
__device__ __forceinline__ void async_cp16(u16* lds_ptr, const u16* gptr) {
    unsigned           loff = (unsigned)(unsigned long long)lds_ptr;
    unsigned long long g    = (unsigned long long)gptr;
    asm volatile("global_load_async_to_lds_b128 %0, %1, off"
                 :: "v"(loff), "v"(g) : "memory");
}
__device__ __forceinline__ void wait_async0() {
    asm volatile("s_wait_asynccnt 0x0" ::: "memory");
}
__device__ __forceinline__ void wait_async8() {     // leave newest tile (8 ops) in flight
    asm volatile("s_wait_asynccnt 0x8" ::: "memory");
}

__device__ __forceinline__ float gelu_tanh(float x) {
    const float k = 0.7978845608028654f;               // sqrt(2/pi)
    float t = tanhf(k * (x + 0.044715f * x * x * x));
    return 0.5f * x * (1.0f + t);
}

// ---------------------------------------------------------------------------
// Weight convert + transpose: in fp32 [K,N] -> out bf16 [N,K].  64x64 tiles,
// both global sides coalesced; transpose goes through LDS.
// ---------------------------------------------------------------------------
__global__ __launch_bounds__(256) void convert_transpose_kernel(const float* __restrict__ in,
                                                                u16* __restrict__ out,
                                                                int K, int N) {
    constexpr int LT = 72;                       // 144B row stride (16B multiple)
    __shared__ u16 Ls[64 * LT];
    const int k0 = blockIdx.x * 64, n0 = blockIdx.y * 64;
    const int tid = threadIdx.x;
    const int kr = tid >> 2, c = (tid & 3) * 16;

    const float* src = in + (size_t)(k0 + kr) * N + n0 + c;
    #pragma unroll
    for (int i = 0; i < 16; ++i) Ls[(c + i) * LT + kr] = f2bf(src[i]);
    __syncthreads();

    u16* dst = out + (size_t)(n0 + kr) * K + k0 + c;   // kr reused as n-row
    const v4u* s = reinterpret_cast<const v4u*>(Ls + kr * LT + c);
    v4u x0 = s[0], x1 = s[1];
    *reinterpret_cast<v4u*>(dst)     = x0;
    *reinterpret_cast<v4u*>(dst + 8) = x1;
}

// ---------------------------------------------------------------------------
// LayerNorm over C=768, one block per row, bf16 output
// ---------------------------------------------------------------------------
__global__ __launch_bounds__(256) void layernorm_bf16_kernel(const float* __restrict__ x,
                                                             const float* __restrict__ w,
                                                             const float* __restrict__ b,
                                                             u16* __restrict__ out, int C) {
    const int row = blockIdx.x, tid = threadIdx.x;
    const float* xr = x + (size_t)row * C;
    __shared__ float red[256];

    float v0 = xr[tid], v1 = xr[tid + 256], v2 = xr[tid + 512];
    red[tid] = v0 + v1 + v2;
    __syncthreads();
    for (int st = 128; st > 0; st >>= 1) { if (tid < st) red[tid] += red[tid + st]; __syncthreads(); }
    float mu = red[0] / (float)C;
    __syncthreads();

    float d0 = v0 - mu, d1 = v1 - mu, d2 = v2 - mu;
    red[tid] = d0 * d0 + d1 * d1 + d2 * d2;
    __syncthreads();
    for (int st = 128; st > 0; st >>= 1) { if (tid < st) red[tid] += red[tid + st]; __syncthreads(); }
    float rs = rsqrtf(red[0] / (float)C + 1e-5f);

    u16* o = out + (size_t)row * C;
    o[tid]       = f2bf(d0 * rs * w[tid]       + b[tid]);
    o[tid + 256] = f2bf(d1 * rs * w[tid + 256] + b[tid + 256]);
    o[tid + 512] = f2bf(d2 * rs * w[tid + 512] + b[tid + 512]);
}

// ---------------------------------------------------------------------------
// Tiled bf16 GEMM: OUT = act(A[M,K] @ Wt[N,K]^T + bias) [+ res]
// 128x128x32 block tile; 128 threads = 4 waves; each wave owns 64x64
// (16 WMMAs per K-step from 8 fragment loads). Double-buffered async-to-LDS:
// tile k+1 is in flight (ASYNCcnt<=8) while tile k feeds the WMMAs.
// ---------------------------------------------------------------------------
template <int ACT, int RES, int OUTBF>
__global__ __launch_bounds__(128) void gemm_bf16_kernel(const u16* __restrict__ A,
                                                        const u16* __restrict__ Wt,
                                                        const float* __restrict__ bias,
                                                        const float* __restrict__ res,
                                                        float* __restrict__ outf,
                                                        u16* __restrict__ outb,
                                                        int M, int N, int K) {
    constexpr int BK = 32, LD = BK + 8;           // 80B row stride (16B multiple)
    constexpr int TILE = 128 * LD;
    __shared__ u16 As[2 * TILE];
    __shared__ u16 Ws[2 * TILE];                  // Ws[n][k]

    const int tid  = threadIdx.x;
    const int lane = tid & 31;
    const int half = lane >> 4;
    const int l16  = lane & 15;
    const int wave = tid >> 5;
    const int wm   = wave >> 1;                   // 0..1 : 64-row strip
    const int wn   = wave & 1;                    // 0..1 : 64-col strip
    const int bm0  = blockIdx.y * 128;
    const int bn0  = blockIdx.x * 128;

    v8f acc[4][4];
    for (int i = 0; i < 4; ++i)
        for (int j = 0; j < 4; ++j)
            for (int r = 0; r < 8; ++r) acc[i][j][r] = 0.0f;

    const int lrow = tid >> 1;                    // 0..63
    const int lc   = (tid & 1) * 16;              // 0 / 16

    // 8 async instructions per wave per tile (A:4 + W:4)
    auto issue_tile = [&](int kk, int bsel) {
        #pragma unroll
        for (int it = 0; it < 2; ++it) {          // rows lrow and lrow+64
            int r = lrow + it * 64;
            const u16* ga = A  + (size_t)(bm0 + r) * K + kk + lc;
            const u16* gw = Wt + (size_t)(bn0 + r) * K + kk + lc;
            u16* la = As + bsel * TILE + r * LD + lc;
            u16* lw = Ws + bsel * TILE + r * LD + lc;
            async_cp16(la,     ga);
            async_cp16(la + 8, ga + 8);
            async_cp16(lw,     gw);
            async_cp16(lw + 8, gw + 8);
        }
    };

    issue_tile(0, 0);
    int buf = 0;
    for (int k0 = 0; k0 < K; k0 += BK, buf ^= 1) {
        const bool more = (k0 + BK) < K;
        if (more) issue_tile(k0 + BK, buf ^ 1);   // prefetch next tile (other buffer)
        if (more) wait_async8(); else wait_async0();
        __syncthreads();

        const u16* Ab = As + buf * TILE;
        const u16* Wb = Ws + buf * TILE;
        Frag16 af[4], bf[4];
        #pragma unroll
        for (int fm = 0; fm < 4; ++fm) {          // A frag: row l16, K halves
            const u16* p = Ab + (wm * 64 + fm * 16 + l16) * LD + half * 8;
            af[fm].q[0] = *reinterpret_cast<const v4u*>(p);
            af[fm].q[1] = *reinterpret_cast<const v4u*>(p + 16);
        }
        #pragma unroll
        for (int fn = 0; fn < 4; ++fn) {          // B frag: col l16, K halves
            const u16* p = Wb + (wn * 64 + fn * 16 + l16) * LD + half * 8;
            bf[fn].q[0] = *reinterpret_cast<const v4u*>(p);
            bf[fn].q[1] = *reinterpret_cast<const v4u*>(p + 16);
        }
        #pragma unroll
        for (int fm = 0; fm < 4; ++fm)
            #pragma unroll
            for (int fn = 0; fn < 4; ++fn)
                acc[fm][fn] = wmma_bf16(af[fm], bf[fn], acc[fm][fn]);
        __syncthreads();
    }

    #pragma unroll
    for (int fm = 0; fm < 4; ++fm)
        #pragma unroll
        for (int fn = 0; fn < 4; ++fn) {
            int col = bn0 + wn * 64 + fn * 16 + l16;
            float bv = bias[col];
            #pragma unroll
            for (int r = 0; r < 8; ++r) {
                int row = bm0 + wm * 64 + fm * 16 + half * 8 + r;
                float v = acc[fm][fn][r] + bv;
                if (ACT == 1) v = gelu_tanh(v);
                if (RES)      v += res[(size_t)row * N + col];
                if (OUTBF) outb[(size_t)row * N + col] = f2bf(v);
                else       outf[(size_t)row * N + col] = v;
            }
        }
}

// ---------------------------------------------------------------------------
// Flash-style causal attention. qkv: bf16 [B*T, 3C] (q|k|v). y: bf16 [B*T, C].
// Grid (T/64, B*NH); 128 threads = 4 waves; each wave owns 16 query rows.
// ---------------------------------------------------------------------------
#define ATT_T   2048
#define ATT_C   768
#define ATT_NH  12
#define ATT_HD  64

__global__ __launch_bounds__(128) void attn_kernel(const u16* __restrict__ qkv,
                                                   u16* __restrict__ y) {
    constexpr int LDK = ATT_HD + 8;      // 72 u16 = 144B, 16B multiple
    __shared__ u16 Kt[64 * LDK];         // K tile  [key][d]
    __shared__ u16 Vt[64 * LDK];         // V tile transposed [d][key]
    __shared__ u16 Pw[4][16 * LDK];      // per-wave P staging [row][key]

    const int bh   = blockIdx.y;
    const int bb   = bh / ATT_NH;
    const int hh   = bh % ATT_NH;
    const int qt   = blockIdx.x;
    const int q0   = qt * 64;
    const int tid  = threadIdx.x;
    const int wave = tid >> 5;
    const int lane = tid & 31;
    const int half = lane >> 4;
    const int l16  = lane & 15;
    const size_t rstride = 3 * ATT_C;    // 2304 elems per token row

    Frag16 qf[2];
    {
        int qrow = q0 + wave * 16 + l16;
        const u16* p = qkv + ((size_t)(bb * ATT_T + qrow)) * rstride + hh * ATT_HD;
        #pragma unroll
        for (int kc = 0; kc < 2; ++kc) {
            const u16* pp = p + kc * 32 + half * 8;
            qf[kc].q[0] = *reinterpret_cast<const v4u*>(pp);
            qf[kc].q[1] = *reinterpret_cast<const v4u*>(pp + 16);
        }
    }

    v8f oa[4];
    for (int fd = 0; fd < 4; ++fd)
        for (int r = 0; r < 8; ++r) oa[fd][r] = 0.0f;
    float mrun[8], lrun[8];
    for (int r = 0; r < 8; ++r) { mrun[r] = -INFINITY; lrun[r] = 0.0f; }
    const float scale = 0.125f;          // 1/sqrt(HD)

    for (int kt = 0; kt <= qt; ++kt) {
        const int kb = kt * 64;
        {   // K tile: async global->LDS, row-major [key][d]
            int krow = tid >> 1, d0 = (tid & 1) * 32;
            const u16* gk = qkv + (size_t)(bb * ATT_T + kb + krow) * rstride
                                + ATT_C + hh * ATT_HD + d0;
            u16* lk = Kt + krow * LDK + d0;
            async_cp16(lk,      gk);
            async_cp16(lk + 8,  gk + 8);
            async_cp16(lk + 16, gk + 16);
            async_cp16(lk + 24, gk + 24);
        }
        {   // V tile: 2 keys x 16 d per thread, packed pair stores (b32)
            int key0 = (tid >> 2) * 2, dv0 = (tid & 3) * 16;
            const u16* vp = qkv + (size_t)(bb * ATT_T + kb + key0) * rstride
                                + 2 * ATT_C + hh * ATT_HD + dv0;
            union { v4u q[2]; u16 u[16]; } a, b;
            a.q[0] = *reinterpret_cast<const v4u*>(vp);
            a.q[1] = *reinterpret_cast<const v4u*>(vp + 8);
            b.q[0] = *reinterpret_cast<const v4u*>(vp + rstride);
            b.q[1] = *reinterpret_cast<const v4u*>(vp + rstride + 8);
            #pragma unroll
            for (int i = 0; i < 16; ++i)
                *reinterpret_cast<unsigned*>(Vt + (dv0 + i) * LDK + key0) =
                    (unsigned)a.u[i] | ((unsigned)b.u[i] << 16);
        }
        wait_async0();
        __syncthreads();

        // S = Q @ K^T
        v8f sa[4];
        #pragma unroll
        for (int fn = 0; fn < 4; ++fn) {
            for (int r = 0; r < 8; ++r) sa[fn][r] = 0.0f;
            #pragma unroll
            for (int kc = 0; kc < 2; ++kc) {
                Frag16 bk;
                const u16* p = Kt + (fn * 16 + l16) * LDK + kc * 32 + half * 8;
                bk.q[0] = *reinterpret_cast<const v4u*>(p);
                bk.q[1] = *reinterpret_cast<const v4u*>(p + 16);
                sa[fn] = wmma_bf16(qf[kc], bk, sa[fn]);
            }
        }
        // scale + causal mask
        #pragma unroll
        for (int fn = 0; fn < 4; ++fn)
            #pragma unroll
            for (int r = 0; r < 8; ++r) {
                float s  = sa[fn][r] * scale;
                int   kj = kb + fn * 16 + l16;
                int   qi = q0 + wave * 16 + half * 8 + r;
                sa[fn][r] = (kj > qi) ? -INFINITY : s;
            }
        // online softmax (row lives in one 16-lane group; xor 1/2/4/8 reduces)
        #pragma unroll
        for (int r = 0; r < 8; ++r) {
            float mt = fmaxf(fmaxf(sa[0][r], sa[1][r]), fmaxf(sa[2][r], sa[3][r]));
            for (int m = 1; m < 16; m <<= 1) mt = fmaxf(mt, __shfl_xor(mt, m, 32));
            float mn = fmaxf(mrun[r], mt);
            float al = __expf(mrun[r] - mn);
            lrun[r] *= al;
            for (int fd = 0; fd < 4; ++fd) oa[fd][r] *= al;
            float rs = 0.0f;
            #pragma unroll
            for (int fn = 0; fn < 4; ++fn) {
                float p = __expf(sa[fn][r] - mn);
                sa[fn][r] = p;
                rs += p;
            }
            for (int m = 1; m < 16; m <<= 1) rs += __shfl_xor(rs, m, 32);
            lrun[r] += rs;
            mrun[r] = mn;
        }
        // stage P (bf16) per-wave: C-layout -> A-fragment layout
        u16* pw = &Pw[wave][0];
        #pragma unroll
        for (int fn = 0; fn < 4; ++fn)
            #pragma unroll
            for (int r = 0; r < 8; ++r)
                pw[(half * 8 + r) * LDK + fn * 16 + l16] = f2bf(sa[fn][r]);
        asm volatile("s_wait_dscnt 0x0" ::: "memory");

        // O += P @ V
        #pragma unroll
        for (int kc = 0; kc < 2; ++kc) {
            Frag16 pa;
            const u16* pp = pw + l16 * LDK + kc * 32 + half * 8;
            pa.q[0] = *reinterpret_cast<const v4u*>(pp);
            pa.q[1] = *reinterpret_cast<const v4u*>(pp + 16);
            #pragma unroll
            for (int fd = 0; fd < 4; ++fd) {
                Frag16 bv;
                const u16* vp2 = Vt + (fd * 16 + l16) * LDK + kc * 32 + half * 8;
                bv.q[0] = *reinterpret_cast<const v4u*>(vp2);
                bv.q[1] = *reinterpret_cast<const v4u*>(vp2 + 16);
                oa[fd] = wmma_bf16(pa, bv, oa[fd]);
            }
        }
        __syncthreads();
    }

    #pragma unroll
    for (int fd = 0; fd < 4; ++fd)
        #pragma unroll
        for (int r = 0; r < 8; ++r) {
            int tokn = q0 + wave * 16 + half * 8 + r;
            float v = oa[fd][r] / lrun[r];
            y[((size_t)(bb * ATT_T + tokn)) * ATT_C + hh * ATT_HD + fd * 16 + l16] = f2bf(v);
        }
}

// ---------------------------------------------------------------------------
// Launcher
// ---------------------------------------------------------------------------
extern "C" void kernel_launch(void* const* d_in, const int* in_sizes, int n_in,
                              void* d_out, int out_size, void* d_ws, size_t ws_size,
                              hipStream_t stream) {
    constexpr int Bn = 2, Tn = 2048, Cn = 768, NHn = 12, DFFn = 3072;
    constexpr int Mn = Bn * Tn;                              // 4096

    const float* x        = (const float*)d_in[0];
    const float* ln1_w    = (const float*)d_in[1];
    const float* ln1_b    = (const float*)d_in[2];
    const float* c_attn_w = (const float*)d_in[3];
    const float* c_attn_b = (const float*)d_in[4];
    const float* c_proj_w = (const float*)d_in[5];
    const float* c_proj_b = (const float*)d_in[6];
    const float* ln2_w    = (const float*)d_in[7];
    const float* ln2_b    = (const float*)d_in[8];
    const float* fc_w     = (const float*)d_in[9];
    const float* fc_b     = (const float*)d_in[10];
    const float* proj_w   = (const float*)d_in[11];
    const float* proj_b   = (const float*)d_in[12];
    float*       out      = (float*)d_out;

    size_t off = 0;
    auto take = [&](size_t bytes) -> void* {
        void* p = (char*)d_ws + off;
        off += (bytes + 255) & ~(size_t)255;
        return p;
    };
    u16*   wq   = (u16*)  take((size_t)Cn * 3 * Cn * 2);     // c_attn_w^T bf16 [3C,C]
    u16*   wp   = (u16*)  take((size_t)Cn * Cn * 2);         // c_proj_w^T bf16 [C,C]
    u16*   wf   = (u16*)  take((size_t)Cn * DFFn * 2);       // fc_w^T bf16 [DFF,C]
    u16*   wo   = (u16*)  take((size_t)DFFn * Cn * 2);       // proj_w^T bf16 [C,DFF]
    u16*   h1   = (u16*)  take((size_t)Mn * Cn * 2);         // ln1(x) bf16
    u16*   qkvb = (u16*)  take((size_t)Mn * 3 * Cn * 2);     // qkv bf16
    u16*   yb   = (u16*)  take((size_t)Mn * Cn * 2);         // attention out bf16
    float* x1   = (float*)take((size_t)Mn * Cn * 4);         // x + attn proj (fp32)
    u16*   h2   = (u16*)  take((size_t)Mn * Cn * 2);         // ln2(x1) bf16
    u16*   actb = (u16*)  take((size_t)Mn * DFFn * 2);       // gelu(fc) bf16

    auto cvtT = [&](const float* src, u16* dst, int K, int N) {
        convert_transpose_kernel<<<dim3(K / 64, N / 64), 256, 0, stream>>>(src, dst, K, N);
    };
    cvtT(c_attn_w, wq, Cn, 3 * Cn);
    cvtT(c_proj_w, wp, Cn, Cn);
    cvtT(fc_w,     wf, Cn, DFFn);
    cvtT(proj_w,   wo, DFFn, Cn);

    // 1. ln1(x) -> h1
    layernorm_bf16_kernel<<<Mn, 256, 0, stream>>>(x, ln1_w, ln1_b, h1, Cn);
    // 2. qkv = h1 @ Wqkv + b         [4096 x 2304]
    gemm_bf16_kernel<0, 0, 1><<<dim3(3 * Cn / 128, Mn / 128), 128, 0, stream>>>(
        h1, wq, c_attn_b, nullptr, nullptr, qkvb, Mn, 3 * Cn, Cn);
    // 3. causal flash attention -> yb
    attn_kernel<<<dim3(Tn / 64, Bn * NHn), 128, 0, stream>>>(qkvb, yb);
    // 4. x1 = x + yb @ Wproj + b     [4096 x 768] fp32
    gemm_bf16_kernel<0, 1, 0><<<dim3(Cn / 128, Mn / 128), 128, 0, stream>>>(
        yb, wp, c_proj_b, x, x1, nullptr, Mn, Cn, Cn);
    // 5. ln2(x1) -> h2
    layernorm_bf16_kernel<<<Mn, 256, 0, stream>>>(x1, ln2_w, ln2_b, h2, Cn);
    // 6. act = gelu(h2 @ Wfc + b)    [4096 x 3072]
    gemm_bf16_kernel<1, 0, 1><<<dim3(DFFn / 128, Mn / 128), 128, 0, stream>>>(
        h2, wf, fc_b, nullptr, nullptr, actb, Mn, DFFn, Cn);
    // 7. out = x1 + act @ Wproj2 + b [4096 x 768] fp32
    gemm_bf16_kernel<0, 1, 0><<<dim3(Cn / 128, Mn / 128), 128, 0, stream>>>(
        actb, wo, proj_b, x1, out, nullptr, Mn, Cn, DFFn);
}
